// GCN_88459146428655
// MI455X (gfx1250) — matrix operand order repaired and checked by reading, not compiled, not run on hardware
//
#include <hip/hip_runtime.h>

typedef __attribute__((ext_vector_type(16))) _Float16 v16h;
typedef __attribute__((ext_vector_type(8)))  float    v8f;

#define NN 50000
#define EE 600000
#define KK 128          // inner dimension for every GEMM in this net
#define KC (KK / 32)    // K-chunks of 32
#define EPSV 1e-5f

// ---------------- elementwise helpers ----------------

__global__ void k_fill(float* __restrict__ p, float val, int n) {
    int i = blockIdx.x * blockDim.x + threadIdx.x;
    if (i < n) p[i] = val;
}

__global__ void k_degcount(const int* __restrict__ col, float* __restrict__ deg, int e) {
    int i = blockIdx.x * blockDim.x + threadIdx.x;
    if (i < e) unsafeAtomicAdd(&deg[col[i]], 1.0f);
}

__global__ void k_rsqrt(float* __restrict__ p, int n) {
    int i = blockIdx.x * blockDim.x + threadIdx.x;
    if (i < n) p[i] = rsqrtf(p[i]);
}

// ---------------- fragment packers (f32 -> f16 in WMMA fragment order) ----------------
// A fragment layout (16-bit A 16x32, ISA 7.12.2): lane l holds row m=l&15;
//   element e -> K = (l>>4)*8 + (e<8 ? e : e+8) within a 32-wide chunk.
// Apack linear index: ((tm*KC + kc)*32 + lane)*16 + e
__global__ void k_packA(const float* __restrict__ X, _Float16* __restrict__ Apack, int total) {
    int idx = blockIdx.x * blockDim.x + threadIdx.x;
    if (idx >= total) return;
    const int e    = idx & 15;
    const int lane = (idx >> 4) & 31;
    const int rest = idx >> 9;              // tm*KC + kc
    const int kc   = rest & (KC - 1);
    const int tm   = rest >> 2;             // KC == 4
    const int row  = tm * 16 + (lane & 15);
    const int k    = kc * 32 + (lane >> 4) * 8 + (e < 8 ? e : e + 8);
    Apack[idx] = (_Float16)X[(size_t)row * KK + k];
}

// B fragment layout (16-bit B 32x16, from sparse-B table halved): lane l holds col n=l&15;
//   element e -> K = (l>>4)*8 + (e<8 ? e : e+8) within a 32-wide chunk.
// Bpack linear index: ((nt*KC + kc)*32 + lane)*16 + e
__global__ void k_packB(const float* __restrict__ W, _Float16* __restrict__ Bpack,
                        int total, int Nout) {
    int idx = blockIdx.x * blockDim.x + threadIdx.x;
    if (idx >= total) return;
    const int e    = idx & 15;
    const int lane = (idx >> 4) & 31;
    const int rest = idx >> 9;              // nt*KC + kc
    const int kc   = rest & (KC - 1);
    const int nt   = rest >> 2;
    const int k    = kc * 32 + (lane >> 4) * 8 + (e < 8 ? e : e + 8);
    const int n    = nt * 16 + (lane & 15);
    Bpack[idx] = (_Float16)W[(size_t)k * Nout + n];
}

// ---------------- WMMA GEMM on packed fragments ----------------
// One wave computes a 16 x (NT*16) output strip: A fragment loaded once per K-chunk,
// reused across NT column tiles. Fully unrolled: KC*NT wmma ops per wave.

template <int NT>
__global__ __launch_bounds__(128) void k_gemm_wmma(const v16h* __restrict__ Apack,
                                                   const v16h* __restrict__ Bpack,
                                                   float* __restrict__ C, int M) {
    const int tiles = M >> 4;
    const int wave  = blockIdx.x * (blockDim.x >> 5) + (threadIdx.x >> 5);
    if (wave >= tiles) return;               // wave-uniform exit: EXEC all-1s for live waves
    const int lane = threadIdx.x & 31;
    const int g    = lane >> 4;
    const int m    = lane & 15;

    const v8f zero = {0.f, 0.f, 0.f, 0.f, 0.f, 0.f, 0.f, 0.f};
    v8f acc[NT];
#pragma unroll
    for (int nt = 0; nt < NT; ++nt) acc[nt] = zero;

#pragma unroll
    for (int kc = 0; kc < KC; ++kc) {
        const v16h a = Apack[(size_t)(wave * KC + kc) * 32 + lane];
#pragma unroll
        for (int nt = 0; nt < NT; ++nt) {
            const v16h b = Bpack[(size_t)(nt * KC + kc) * 32 + lane];
            acc[nt] = __builtin_amdgcn_wmma_f32_16x16x32_f16(
                false, a, false, b, (short)0, acc[nt], false, false);
        }
    }

    // D layout: element v -> row wave*16 + g*8 + v, col nt*16 + m
    const int Nout = NT * 16;
    float* Crow = C + (size_t)(wave * 16 + g * 8) * Nout + m;
#pragma unroll
    for (int nt = 0; nt < NT; ++nt)
#pragma unroll
        for (int v = 0; v < 8; ++v)
            Crow[(size_t)v * Nout + nt * 16] = acc[nt][v];
}

// ---------------- self-loop + bias init: agg[i,:] = b + h[i,:] * dis[i]^2 ----------------

__global__ void k_selfbias(const float* __restrict__ h, const float* __restrict__ dis,
                           const float* __restrict__ bias, float* __restrict__ outp,
                           int total, int F) {
    int i = blockIdx.x * blockDim.x + threadIdx.x;
    if (i >= total) return;
    const int F4   = F >> 2;
    const int node = i / F4;
    const int j    = i - node * F4;
    const float d  = dis[node];
    const float w  = d * d;
    const float4 hv = ((const float4*)(h + (size_t)node * F))[j];
    const float4 bv = ((const float4*)bias)[j];
    float4 o;
    o.x = bv.x + hv.x * w;
    o.y = bv.y + hv.y * w;
    o.z = bv.z + hv.z * w;
    o.w = bv.w + hv.w * w;
    ((float4*)(outp + (size_t)node * F))[j] = o;
}

// ---------------- edge scatter-add: agg[col,:] += h[row,:] * dis[row]*dis[col] ----------------

__global__ void k_edgeagg(const int* __restrict__ row, const int* __restrict__ col,
                          const float* __restrict__ dis, const float* __restrict__ h,
                          float* __restrict__ outp, int total, int F) {
    int i = blockIdx.x * blockDim.x + threadIdx.x;
    if (i >= total) return;
    const int F4 = F >> 2;
    const int e  = i / F4;
    const int j  = i - e * F4;
    const int r  = row[e];
    const int c  = col[e];
    const float w = dis[r] * dis[c];
    const float4 hv = ((const float4*)(h + (size_t)r * F))[j];
    float* dst = outp + (size_t)c * F + (size_t)j * 4;
    unsafeAtomicAdd(dst + 0, hv.x * w);
    unsafeAtomicAdd(dst + 1, hv.y * w);
    unsafeAtomicAdd(dst + 2, hv.z * w);
    unsafeAtomicAdd(dst + 3, hv.w * w);
}

// ---------------- BN + ReLU (inference), in place ----------------

__global__ void k_bnrelu(float* __restrict__ a, const float* __restrict__ g,
                         const float* __restrict__ bt, const float* __restrict__ m,
                         const float* __restrict__ v, int n, int F) {
    int i = blockIdx.x * blockDim.x + threadIdx.x;
    if (i >= n) return;
    const int f = i & (F - 1);
    float val = (a[i] - m[f]) * rsqrtf(v[f] + EPSV) * g[f] + bt[f];
    a[i] = val > 0.f ? val : 0.f;
}

// ---------------- launcher ----------------

extern "C" void kernel_launch(void* const* d_in, const int* in_sizes, int n_in,
                              void* d_out, int out_size, void* d_ws, size_t ws_size,
                              hipStream_t stream) {
    (void)in_sizes; (void)n_in; (void)out_size; (void)ws_size;

    const float* x   = (const float*)d_in[0];
    const int*   ei  = (const int*)d_in[1];
    const float* W1  = (const float*)d_in[2];
    const float* b1  = (const float*)d_in[3];
    const float* g1  = (const float*)d_in[4];
    const float* bt1 = (const float*)d_in[5];
    const float* m1  = (const float*)d_in[6];
    const float* v1  = (const float*)d_in[7];
    const float* W2  = (const float*)d_in[8];
    const float* b2  = (const float*)d_in[9];
    const float* g2  = (const float*)d_in[10];
    const float* bt2 = (const float*)d_in[11];
    const float* m2  = (const float*)d_in[12];
    const float* v2  = (const float*)d_in[13];
    const float* W3  = (const float*)d_in[14];
    const float* b3  = (const float*)d_in[15];
    float* out = (float*)d_out;

    const int* rowp = ei;        // sources
    const int* colp = ei + EE;   // targets

    // workspace carve (256B aligned)
    char* p = (char*)d_ws;
    auto carve = [&](size_t bytes) {
        void* r = (void*)p;
        p += (bytes + 255) & ~(size_t)255;
        return r;
    };
    float*    dis   = (float*)carve((size_t)NN * 4);                 // deg -> d^-1/2
    _Float16* Apack = (_Float16*)carve((size_t)NN * KK * 2);         // packed A fragments
    _Float16* Bpack = (_Float16*)carve((size_t)KK * 128 * 2);        // packed B fragments
    float*    h     = (float*)carve((size_t)NN * 128 * 4);           // GEMM output
    float*    agg   = (float*)carve((size_t)NN * 128 * 4);           // aggregated output

    auto cdiv = [](long a, long b) { return (int)((a + b - 1) / b); };

    // ---- normalization: deg = 1 (self loop) + indegree; dis = deg^-1/2 ----
    k_fill<<<cdiv(NN, 256), 256, 0, stream>>>(dis, 1.0f, NN);
    k_degcount<<<cdiv(EE, 256), 256, 0, stream>>>(colp, dis, EE);
    k_rsqrt<<<cdiv(NN, 256), 256, 0, stream>>>(dis, NN);

    const int gemmBlocks = cdiv((long)(NN / 16) * 32, 128);

    // ---- layer 1 ----
    k_packA<<<cdiv((long)NN * KK, 256), 256, 0, stream>>>(x, Apack, NN * KK);
    k_packB<<<cdiv(KK * 128, 256), 256, 0, stream>>>(W1, Bpack, KK * 128, 128);
    k_gemm_wmma<8><<<gemmBlocks, 128, 0, stream>>>((const v16h*)Apack, (const v16h*)Bpack, h, NN);
    k_selfbias<<<cdiv((long)NN * 32, 256), 256, 0, stream>>>(h, dis, b1, agg, NN * 32, 128);
    k_edgeagg<<<cdiv((long)EE * 32, 256), 256, 0, stream>>>(rowp, colp, dis, h, agg, EE * 32, 128);
    k_bnrelu<<<cdiv((long)NN * 128, 256), 256, 0, stream>>>(agg, g1, bt1, m1, v1, NN * 128, 128);

    // ---- layer 2 ----
    k_packA<<<cdiv((long)NN * KK, 256), 256, 0, stream>>>(agg, Apack, NN * KK);
    k_packB<<<cdiv(KK * 128, 256), 256, 0, stream>>>(W2, Bpack, KK * 128, 128);
    k_gemm_wmma<8><<<gemmBlocks, 128, 0, stream>>>((const v16h*)Apack, (const v16h*)Bpack, h, NN);
    k_selfbias<<<cdiv((long)NN * 32, 256), 256, 0, stream>>>(h, dis, b2, agg, NN * 32, 128);
    k_edgeagg<<<cdiv((long)EE * 32, 256), 256, 0, stream>>>(rowp, colp, dis, h, agg, EE * 32, 128);
    k_bnrelu<<<cdiv((long)NN * 128, 256), 256, 0, stream>>>(agg, g2, bt2, m2, v2, NN * 128, 128);

    // ---- layer 3 (F_out = 64, straight into d_out) ----
    k_packA<<<cdiv((long)NN * KK, 256), 256, 0, stream>>>(agg, Apack, NN * KK);
    k_packB<<<cdiv(KK * 64, 256), 256, 0, stream>>>(W3, Bpack, KK * 64, 64);
    k_gemm_wmma<4><<<gemmBlocks, 128, 0, stream>>>((const v16h*)Apack, (const v16h*)Bpack, h, NN);
    k_selfbias<<<cdiv((long)NN * 16, 256), 256, 0, stream>>>(h, dis, b3, out, NN * 16, 64);
    k_edgeagg<<<cdiv((long)EE * 16, 256), 256, 0, stream>>>(rowp, colp, dis, h, out, EE * 16, 64);
}
